// CNN_tcn_81432579932430
// MI455X (gfx1250) — compile-verified, matrix operand order checked by reading, not run on hardware
//
#include <hip/hip_runtime.h>
#include <math.h>

#define ADIM 512
#define DDIM 64
#define EPSV 1e-5f

typedef __attribute__((ext_vector_type(16))) __bf16 v16bf;
typedef __attribute__((ext_vector_type(8)))  float  v8f;

__device__ __forceinline__ float geluf(float x) {
    return 0.5f * x * (1.0f + erff(x * 0.7071067811865476f));
}

__device__ __forceinline__ unsigned short f2bf(float x) {
    unsigned int u = __float_as_uint(x);
    unsigned int r = u + 0x7FFFu + ((u >> 16) & 1u);   // round-to-nearest-even
    return (unsigned short)(r >> 16);
}
__device__ __forceinline__ float bf2f(unsigned short h) {
    return __uint_as_float(((unsigned int)h) << 16);
}

// Dilated (1,3) conv, dilation 2. Row stored with +2 zero-pad on each side:
// row[ci*68 + (d_in + 2)]. Output at d reads stored idx d, d+2, d+4.
__device__ __forceinline__ void conv3tap(const float* row, int d, const float* w, float out[4]) {
#pragma unroll
    for (int co = 0; co < 4; ++co) {
        float acc = 0.f;
#pragma unroll
        for (int ci = 0; ci < 4; ++ci) {
            const float* wr = w + co * 12 + ci * 3;
            const float* rr = row + ci * 68 + d;
            acc += rr[0] * wr[0] + rr[2] * wr[1] + rr[4] * wr[2];
        }
        out[co] = acc;
    }
}

// normalize (affine LN via precomputed m,inv), 4x4 pointwise, exact GELU
__device__ __forceinline__ void pwgelu(const float c[4], float m, float inv,
                                       const float* pw, float out[4]) {
    float n0 = (c[0] - m) * inv, n1 = (c[1] - m) * inv;
    float n2 = (c[2] - m) * inv, n3 = (c[3] - m) * inv;
#pragma unroll
    for (int o = 0; o < 4; ++o) {
        float a = pw[o * 4 + 0] * n0 + pw[o * 4 + 1] * n1 +
                  pw[o * 4 + 2] * n2 + pw[o * 4 + 3] * n3;
        out[o] = geluf(a);
    }
}

__device__ __forceinline__ void lnStats(const float* st, int b, float N, float& m, float& inv) {
    float s = st[2 * b], q = st[2 * b + 1];
    m = s / N;
    float v = q / N - m * m;
    inv = rsqrtf(v + EPSV);
}

// deterministic fixed-order tree reduction over 256 threads, 2 values at once
__device__ __forceinline__ void blockReduce2(float& a, float& b, float* r0, float* r1) {
    int t = threadIdx.x;
    __syncthreads();
    r0[t] = a; r1[t] = b;
    __syncthreads();
    for (int s = 128; s > 0; s >>= 1) {
        if (t < s) { r0[t] += r0[t + s]; r1[t] += r1[t + s]; }
        __syncthreads();
    }
    a = r0[0]; b = r1[0];
}

// ---------------------------------------------------------------------------
// K0: fold cw (4x12) through c2w (16x4x64) -> W2[16][768] bf16, S[o2]=sum(W2 row)
// ---------------------------------------------------------------------------
__global__ __launch_bounds__(256) void prep_kernel(const float* __restrict__ cw,
                                                   const float* __restrict__ c2w,
                                                   unsigned short* __restrict__ W2bf,
                                                   float* __restrict__ S) {
    const int t = threadIdx.x;
    __shared__ float w2tmp[12288];
    for (int i = t; i < 12288; i += 256) {
        int o2 = i / 768, K = i % 768, k = K / 64, dd = K % 64;
        float v = 0.f;
#pragma unroll
        for (int c = 0; c < 4; ++c)
            v += c2w[o2 * 256 + c * 64 + dd] * cw[c * 12 + k];
        w2tmp[i] = v;
        W2bf[i] = f2bf(v);
    }
    __syncthreads();
    if (t < 16) {
        float acc = 0.f;
        for (int j = 0; j < 768; ++j) acc += bf2f(f2bf(w2tmp[t * 768 + j]));
        S[t] = acc;
    }
}

// ---------------------------------------------------------------------------
// K1..K3: streaming stats passes, one block per batch sample (256 thr = 4 rows x 64 d)
//  STAGE1: conv1 stats
//  STAGE2: x1 stats (concat LN) + conv2 stats
//  STAGE3: x2 stats (concat LN) + conv3 stats
// ---------------------------------------------------------------------------
template <int STAGE>
__global__ __launch_bounds__(256) void stats_kernel(const float* __restrict__ s,
                                                    const float* __restrict__ d1w1,
                                                    const float* __restrict__ d1w2,
                                                    const float* __restrict__ d2w1,
                                                    const float* __restrict__ d2w2,
                                                    const float* __restrict__ d3w1,
                                                    const float* __restrict__ st1,
                                                    const float* __restrict__ st2,
                                                    float* __restrict__ stOut,
                                                    float* __restrict__ sc) {
    const int b = blockIdx.x;
    const int t = threadIdx.x;
    const int r = t >> 6;     // row within group of 4
    const int d = t & 63;

    __shared__ float trow[4 * 4 * 68];     // [r][c][68] padded rows
    __shared__ float wsm[176];             // d1w1|d1w2|d2w1|d2w2|d3w1
    __shared__ float red0[256], red1[256];

    for (int i = t; i < 176; i += 256) {
        float v;
        if (i < 48)       v = d1w1[i];
        else if (i < 64)  v = d1w2[i - 48];
        else if (i < 112) v = d2w1[i - 64];
        else if (i < 128) v = d2w2[i - 112];
        else              v = d3w1[i - 128];
        wsm[i] = v;
    }
    if (t < 64) {  // zero halo pads once
        int rr = t >> 4, cc = (t >> 2) & 3, p = t & 3;
        trow[rr * 272 + cc * 68 + ((p < 2) ? p : (64 + p))] = 0.f;
    }
    __syncthreads();

    float m1 = 0.f, i1v = 1.f, m2 = 0.f, i2v = 1.f;
    if (STAGE >= 2) lnStats(st1, b, 131072.f, m1, i1v);
    if (STAGE >= 3) lnStats(st2, b, 131072.f, m2, i2v);

    float aS = 0.f, aQ = 0.f, xS = 0.f, xQ = 0.f;
    float* rowp = trow + r * 272;

    for (int it = 0; it < 128; ++it) {
        const int a = it * 4 + r;
        const float4 sv = ((const float4*)s)[(size_t)(b * ADIM + a) * DDIM + d];
        __syncthreads();
        rowp[0 * 68 + d + 2] = sv.x;  rowp[1 * 68 + d + 2] = sv.y;
        rowp[2 * 68 + d + 2] = sv.z;  rowp[3 * 68 + d + 2] = sv.w;
        __syncthreads();
        float c1[4]; conv3tap(rowp, d, wsm, c1);
        if (STAGE == 1) {
            aS += c1[0] + c1[1] + c1[2] + c1[3];
            aQ += c1[0]*c1[0] + c1[1]*c1[1] + c1[2]*c1[2] + c1[3]*c1[3];
            continue;
        }
        float x1[4]; pwgelu(c1, m1, i1v, wsm + 48, x1);
        float t2a[4] = { sv.x + x1[0], sv.y + x1[1], sv.z + x1[2], sv.w + x1[3] };
        if (STAGE == 2) {
            xS += x1[0] + x1[1] + x1[2] + x1[3];
            xQ += x1[0]*x1[0] + x1[1]*x1[1] + x1[2]*x1[2] + x1[3]*x1[3];
        }
        __syncthreads();
        rowp[0 * 68 + d + 2] = t2a[0]; rowp[1 * 68 + d + 2] = t2a[1];
        rowp[2 * 68 + d + 2] = t2a[2]; rowp[3 * 68 + d + 2] = t2a[3];
        __syncthreads();
        float c2[4]; conv3tap(rowp, d, wsm + 64, c2);
        if (STAGE == 2) {
            aS += c2[0] + c2[1] + c2[2] + c2[3];
            aQ += c2[0]*c2[0] + c2[1]*c2[1] + c2[2]*c2[2] + c2[3]*c2[3];
            continue;
        }
        float x2[4]; pwgelu(c2, m2, i2v, wsm + 112, x2);
        xS += x2[0] + x2[1] + x2[2] + x2[3];
        xQ += x2[0]*x2[0] + x2[1]*x2[1] + x2[2]*x2[2] + x2[3]*x2[3];
        float t3a[4] = { t2a[0] + x2[0], t2a[1] + x2[1], t2a[2] + x2[2], t2a[3] + x2[3] };
        __syncthreads();
        rowp[0 * 68 + d + 2] = t3a[0]; rowp[1 * 68 + d + 2] = t3a[1];
        rowp[2 * 68 + d + 2] = t3a[2]; rowp[3 * 68 + d + 2] = t3a[3];
        __syncthreads();
        float c3[4]; conv3tap(rowp, d, wsm + 128, c3);
        aS += c3[0] + c3[1] + c3[2] + c3[3];
        aQ += c3[0]*c3[0] + c3[1]*c3[1] + c3[2]*c3[2] + c3[3]*c3[3];
    }

    blockReduce2(aS, aQ, red0, red1);
    if (t == 0) { stOut[2 * b] = aS; stOut[2 * b + 1] = aQ; }
    if (STAGE >= 2) {
        blockReduce2(xS, xQ, red0, red1);
        if (t == 0) {
            if (STAGE == 2) { sc[2 * b] = xS;  sc[2 * b + 1] = xQ; }
            else            { sc[2 * b] += xS; sc[2 * b + 1] += xQ; }
        }
    }
}

// ---------------------------------------------------------------------------
// K4: recompute x1/x2/x3 per 16-row tile, stage bf16 A-tile in LDS, then
//     16x16x768 GEMM via 24x v_wmma_f32_16x16x32_bf16 -> zraw[b][a][16].
//     One wave per block; grid = (8 a-chunks, B).
// ---------------------------------------------------------------------------
__global__ __launch_bounds__(32) void gemm_kernel(const float* __restrict__ s,
                                                  const float* __restrict__ d1w1,
                                                  const float* __restrict__ d1w2,
                                                  const float* __restrict__ d2w1,
                                                  const float* __restrict__ d2w2,
                                                  const float* __restrict__ d3w1,
                                                  const float* __restrict__ d3w2,
                                                  const unsigned short* __restrict__ W2bf,
                                                  const float* __restrict__ st1,
                                                  const float* __restrict__ st2,
                                                  const float* __restrict__ st3,
                                                  float* __restrict__ zraw,
                                                  float* __restrict__ x3part) {
    const int b = blockIdx.y;
    const int chunk = blockIdx.x;
    const int l = threadIdx.x;
    const int d0 = l, d1 = l + 32;

    __shared__ float rb[4 * 68];                 // padded staging row
    __shared__ unsigned short xtile[16 * 768];   // A-tile: 16 rows x K=768, bf16
    __shared__ float wsm[192];

    for (int i = l; i < 192; i += 32) {
        float v;
        if (i < 48)       v = d1w1[i];
        else if (i < 64)  v = d1w2[i - 48];
        else if (i < 112) v = d2w1[i - 64];
        else if (i < 128) v = d2w2[i - 112];
        else if (i < 176) v = d3w1[i - 128];
        else              v = d3w2[i - 176];
        wsm[i] = v;
    }
    if (l < 16) {
        int cc = l >> 2, p = l & 3;
        rb[cc * 68 + ((p < 2) ? p : (64 + p))] = 0.f;
    }
    __syncthreads();

    float m1, i1v, m2, i2v, m3, i3v;
    lnStats(st1, b, 131072.f, m1, i1v);
    lnStats(st2, b, 131072.f, m2, i2v);
    lnStats(st3, b, 131072.f, m3, i3v);

    float sx3 = 0.f, sq3 = 0.f;
    const int m = l & 15;    // A: row M,  B: col N   (lane % 16)
    const int h = l >> 4;    // lane half selects K sub-block

    for (int ti = 0; ti < 4; ++ti) {
        const int abase = (chunk * 4 + ti) * 16;
        for (int r = 0; r < 16; ++r) {
            const int a = abase + r;
            const float4* sp = (const float4*)s + (size_t)(b * ADIM + a) * DDIM;
            const float4 sA = sp[d0];
            const float4 sB = sp[d1];
            __syncthreads();
            rb[0*68 + d0 + 2] = sA.x; rb[1*68 + d0 + 2] = sA.y;
            rb[2*68 + d0 + 2] = sA.z; rb[3*68 + d0 + 2] = sA.w;
            rb[0*68 + d1 + 2] = sB.x; rb[1*68 + d1 + 2] = sB.y;
            rb[2*68 + d1 + 2] = sB.z; rb[3*68 + d1 + 2] = sB.w;
            __syncthreads();
            float c1A[4], c1B[4], x1A[4], x1B[4];
            conv3tap(rb, d0, wsm, c1A);  conv3tap(rb, d1, wsm, c1B);
            pwgelu(c1A, m1, i1v, wsm + 48, x1A);
            pwgelu(c1B, m1, i1v, wsm + 48, x1B);
            float t2A[4] = { sA.x + x1A[0], sA.y + x1A[1], sA.z + x1A[2], sA.w + x1A[3] };
            float t2B[4] = { sB.x + x1B[0], sB.y + x1B[1], sB.z + x1B[2], sB.w + x1B[3] };
            __syncthreads();
            for (int c = 0; c < 4; ++c) { rb[c*68 + d0 + 2] = t2A[c]; rb[c*68 + d1 + 2] = t2B[c]; }
            __syncthreads();
            float c2A[4], c2B[4], x2A[4], x2B[4];
            conv3tap(rb, d0, wsm + 64, c2A);  conv3tap(rb, d1, wsm + 64, c2B);
            pwgelu(c2A, m2, i2v, wsm + 112, x2A);
            pwgelu(c2B, m2, i2v, wsm + 112, x2B);
            float t3A[4] = { t2A[0] + x2A[0], t2A[1] + x2A[1], t2A[2] + x2A[2], t2A[3] + x2A[3] };
            float t3B[4] = { t2B[0] + x2B[0], t2B[1] + x2B[1], t2B[2] + x2B[2], t2B[3] + x2B[3] };
            __syncthreads();
            for (int c = 0; c < 4; ++c) { rb[c*68 + d0 + 2] = t3A[c]; rb[c*68 + d1 + 2] = t3B[c]; }
            __syncthreads();
            float c3A[4], c3B[4], x3A[4], x3B[4];
            conv3tap(rb, d0, wsm + 128, c3A);  conv3tap(rb, d1, wsm + 128, c3B);
            pwgelu(c3A, m3, i3v, wsm + 176, x3A);
            pwgelu(c3B, m3, i3v, wsm + 176, x3B);
#pragma unroll
            for (int c = 0; c < 4; ++c) {
                sx3 += x3A[c] + x3B[c];
                sq3 += x3A[c] * x3A[c] + x3B[c] * x3B[c];
                xtile[r * 768 + (0 + c) * 64 + d0] = f2bf(x1A[c]);
                xtile[r * 768 + (4 + c) * 64 + d0] = f2bf(x2A[c]);
                xtile[r * 768 + (8 + c) * 64 + d0] = f2bf(x3A[c]);
                xtile[r * 768 + (0 + c) * 64 + d1] = f2bf(x1B[c]);
                xtile[r * 768 + (4 + c) * 64 + d1] = f2bf(x2B[c]);
                xtile[r * 768 + (8 + c) * 64 + d1] = f2bf(x3B[c]);
            }
        }
        __syncthreads();

        // ---- 16x16x768 GEMM: 24 chained v_wmma_f32_16x16x32_bf16 ----
        union AccU { v8f v; float f[8]; } cf;
#pragma unroll
        for (int i = 0; i < 8; ++i) cf.f[i] = 0.f;
        for (int ch = 0; ch < 24; ++ch) {
            const int K0 = ch * 32;
            union FragU { v16bf v; unsigned int u[8]; } af, bfv;
#pragma unroll
            for (int j = 0; j < 8; ++j) {
                // 16-bit fragment K mapping (pairs): VGPR j, lane-half h
                const int kl = (j < 4) ? (8 * h + 2 * j) : (16 + 8 * h + 2 * (j - 4));
                af.u[j]  = *(const unsigned int*)&xtile[m * 768 + K0 + kl];
                bfv.u[j] = *(const unsigned int*)&W2bf[m * 768 + K0 + kl];
            }
            cf.v = __builtin_amdgcn_wmma_f32_16x16x32_bf16(
                false, af.v, false, bfv.v, (short)0, cf.v, false, false);
        }
        // C/D layout: VGPR v -> M = v + 8*h ; N = lane%16
#pragma unroll
        for (int vv = 0; vv < 8; ++vv) {
            const int arow = abase + vv + 8 * h;
            zraw[((size_t)(b * ADIM) + arow) * 16 + m] = cf.f[vv];
        }
        __syncthreads();
    }

    // wave-level reduction of x3 stats (wave32)
#pragma unroll
    for (int off = 16; off > 0; off >>= 1) {
        sx3 += __shfl_down(sx3, off);
        sq3 += __shfl_down(sq3, off);
    }
    if (l == 0) {
        x3part[(b * 8 + chunk) * 2 + 0] = sx3;
        x3part[(b * 8 + chunk) * 2 + 1] = sq3;
    }
}

// ---------------------------------------------------------------------------
// K5: apply folded concat-LN correction, LN over [16,A], concat w, 17->1, LN over A
// ---------------------------------------------------------------------------
__global__ __launch_bounds__(256) void final_kernel(const float* __restrict__ zraw,
                                                    const float* __restrict__ x3part,
                                                    const float* __restrict__ sc,
                                                    const float* __restrict__ S,
                                                    const float* __restrict__ c3w,
                                                    const float* __restrict__ w,
                                                    float* __restrict__ out) {
    const int b = blockIdx.x;
    const int t = threadIdx.x;
    __shared__ float zbuf[16 * 512];
    __shared__ float outbuf[512];
    __shared__ float red0[256], red1[256];

    float scS = sc[2 * b], scQ = sc[2 * b + 1];
    for (int j = 0; j < 8; ++j) {
        scS += x3part[(b * 8 + j) * 2 + 0];
        scQ += x3part[(b * 8 + j) * 2 + 1];
    }
    const float Nc = 12.f * 512.f * 64.f;
    const float mc = scS / Nc;
    const float ic = rsqrtf(scQ / Nc - mc * mc + EPSV);

    float zs = 0.f, zq = 0.f;
    for (int i = t; i < 8192; i += 256) {
        const int o2 = i & 15;
        const float v = ic * zraw[(size_t)b * 8192 + i] - ic * mc * S[o2];
        zbuf[i] = v;
        zs += v; zq += v * v;
    }
    blockReduce2(zs, zq, red0, red1);
    const float mz = zs / 8192.f;
    const float iz = rsqrtf(zq / 8192.f - mz * mz + EPSV);
    __syncthreads();

    float os = 0.f, oq = 0.f;
    for (int a = t; a < 512; a += 256) {
        float acc = c3w[16] * w[b * 512 + a];
#pragma unroll
        for (int o = 0; o < 16; ++o)
            acc += c3w[o] * ((zbuf[a * 16 + o] - mz) * iz);
        outbuf[a] = acc;
        os += acc; oq += acc * acc;
    }
    blockReduce2(os, oq, red0, red1);
    const float mo = os / 512.f;
    const float io = rsqrtf(oq / 512.f - mo * mo + EPSV);
    __syncthreads();
    for (int a = t; a < 512; a += 256)
        out[b * 512 + a] = (outbuf[a] - mo) * io;
}

// ---------------------------------------------------------------------------
extern "C" void kernel_launch(void* const* d_in, const int* in_sizes, int n_in,
                              void* d_out, int out_size, void* d_ws, size_t ws_size,
                              hipStream_t stream) {
    const float* s    = (const float*)d_in[0];
    const float* w    = (const float*)d_in[1];
    const float* d1w1 = (const float*)d_in[2];
    const float* d1w2 = (const float*)d_in[3];
    const float* d2w1 = (const float*)d_in[4];
    const float* d2w2 = (const float*)d_in[5];
    const float* d3w1 = (const float*)d_in[6];
    const float* d3w2 = (const float*)d_in[7];
    const float* cw   = (const float*)d_in[8];
    const float* c2w  = (const float*)d_in[9];
    const float* c3w  = (const float*)d_in[10];
    float* out = (float*)d_out;

    // workspace layout (all offsets 64B aligned; total ~8.44 MB)
    char* ws = (char*)d_ws;
    unsigned short* W2bf = (unsigned short*)(ws + 0);      // 16*768*2 = 24576
    float* S    = (float*)(ws + 24576);                    // 16 f32
    float* st1  = (float*)(ws + 24640);                    // 256*2 f32
    float* st2  = (float*)(ws + 26688);
    float* st3  = (float*)(ws + 28736);
    float* sc   = (float*)(ws + 30784);
    float* x3p  = (float*)(ws + 32832);                    // 256*8*2 f32
    float* zraw = (float*)(ws + 49216);                    // 256*512*16 f32

    prep_kernel<<<1, 256, 0, stream>>>(cw, c2w, W2bf, S);
    stats_kernel<1><<<256, 256, 0, stream>>>(s, d1w1, d1w2, d2w1, d2w2, d3w1,
                                             st1, st2, st1, sc);
    stats_kernel<2><<<256, 256, 0, stream>>>(s, d1w1, d1w2, d2w1, d2w2, d3w1,
                                             st1, st2, st2, sc);
    stats_kernel<3><<<256, 256, 0, stream>>>(s, d1w1, d1w2, d2w1, d2w2, d3w1,
                                             st1, st2, st3, sc);
    gemm_kernel<<<dim3(8, 256), 32, 0, stream>>>(s, d1w1, d1w2, d2w1, d2w2, d3w1, d3w2,
                                                 W2bf, st1, st2, st3, zraw, x3p);
    final_kernel<<<256, 256, 0, stream>>>(zraw, x3p, sc, S, c3w, w, out);
}
